// AlternatingDoubleDS_3393024163879
// MI455X (gfx1250) — matrix operand order repaired and checked by reading, not compiled, not run on hardware
//
#include <hip/hip_runtime.h>

// ---------------------------------------------------------------------------
// Types for CDNA5 WMMA (gfx1250, wave32)
// ---------------------------------------------------------------------------
typedef __bf16 bf16;
typedef bf16     v16bf __attribute__((ext_vector_type(16)));
typedef float    v8f   __attribute__((ext_vector_type(8)));
typedef unsigned u32x4 __attribute__((ext_vector_type(4)));

union FragA { v16bf v; unsigned u32[8]; u32x4 q[2]; };
union FragB { v16bf v; u32x4 q[2]; };

__device__ inline unsigned short f2bf(float f) {
  union { float f; unsigned u; } c; c.f = f;
  unsigned u = c.u + 0x7fffu + ((c.u >> 16) & 1u);   // round-to-nearest-even
  return (unsigned short)(u >> 16);
}

// ---------------------------------------------------------------------------
// Core: one wave computes TPW 16x16 output tiles of  C = A[Mx256] * W[256,N]
//   act32 : LDS activations, bf16, rows x 256 cols, viewed as uint pairs
//   Wp    : weights pre-packed in B-fragment order:
//           Wp[(((ct*8)+s)*32 + lane)*16 + e] = bf16(W[s*32 + (e>>3)*16 + (lane>>4)*8 + (e&7)][ct*16 + (lane&15)])
// ---------------------------------------------------------------------------
template<int TPW>
__device__ inline void gemm_compute(const unsigned* act32,
                                    const unsigned short* __restrict__ Wp,
                                    int rb, int lane, int ctBase, v8f* acc) {
  const int half = lane >> 4;
  const unsigned* arow = act32 + (rb * 16 + (lane & 15)) * 128;
#pragma unroll
  for (int s = 0; s < 8; ++s) {
    FragA a;
    // A layout (16-bit, 16x32): VGPR v holds K pair (v&3)*2 + (v>=4?16:0) + half*8
    a.q[0] = *(const u32x4*)(arow + s * 16 + half * 4);
    a.q[1] = *(const u32x4*)(arow + s * 16 + 8 + half * 4);
#pragma unroll
    for (int j = 0; j < TPW; ++j) {
      FragB b;
      const u32x4* wq = (const u32x4*)(Wp + (size_t)(((ctBase + j) * 8 + s) * 32 + lane) * 16);
      b.q[0] = wq[0];
      b.q[1] = wq[1];
      acc[j] = __builtin_amdgcn_wmma_f32_16x16x32_bf16(
          false, a.v, false, b.v, (short)0, acc[j], false, false);
    }
  }
}

template<int TPW>
__device__ inline void zero_acc(v8f* acc) {
#pragma unroll
  for (int j = 0; j < TPW; ++j)
#pragma unroll
    for (int r = 0; r < 8; ++r) acc[j][r] = 0.0f;
}

// bias + (relu) + convert to bf16, store back to LDS activation buffer
template<int TPW, int NOUT, bool RELU>
__device__ inline void store_lds(const v8f* acc, const float* __restrict__ bias,
                                 unsigned short* actOut, int rb, int lane, int ctBase) {
  const int half = lane >> 4;
#pragma unroll
  for (int j = 0; j < TPW; ++j) {
    const int col = (ctBase + j) * 16 + (lane & 15);
    const float bv = bias[col];
#pragma unroll
    for (int r = 0; r < 8; ++r) {
      const int row = rb * 16 + r + half * 8;
      float v = acc[j][r] + bv;
      if (RELU) v = v > 0.0f ? v : 0.0f;
      actOut[row * NOUT + col] = f2bf(v);
    }
  }
}

struct MlpW3 {
  const unsigned short *w0, *w1, *w2;
  const float *b0, *b1, *b2;
};

// ---------------------------------------------------------------------------
// Weight packing: fp32 row-major W[256][N] -> bf16 fragment order
// grid = N blocks x 256 threads  (total 256*N elements)
// ---------------------------------------------------------------------------
__global__ __launch_bounds__(256) void ds_pack_w(const float* __restrict__ W,
                                                 unsigned short* __restrict__ Wp, int N) {
  const int idx = blockIdx.x * 256 + threadIdx.x;
  if (idx >= 256 * N) return;
  const int e    = idx & 15;
  const int lane = (idx >> 4) & 31;
  const int s    = (idx >> 9) & 7;
  const int ct   = idx >> 12;
  const int n = ct * 16 + (lane & 15);
  const int k = s * 32 + (e >> 3) * 16 + (lane >> 4) * 8 + (e & 7);
  Wp[idx] = f2bf(W[k * N + n]);
}

// ---------------------------------------------------------------------------
// DeepSet embedding MLP (256->256->256->256) + segment-sum pooling.
// One workgroup = 128 node rows, 512 threads = 16 waves:
//   wave w -> row-block (w&7), column-half (w>>3); 8 tiles / wave
// This keeps the round-1 per-wave register profile (pipelineable) while
// halving L2 weight traffic vs 64-row tiles.
// LDS = exactly 64KB; pooling buffer overlays dead activations.
// ---------------------------------------------------------------------------
__global__ __launch_bounds__(512) void ds_emb_pool(
    const float* __restrict__ x, const float* __restrict__ ug,
    const int* __restrict__ batchGather, const int* __restrict__ batchSeg,
    MlpW3 w, float* __restrict__ pooled) {
  __shared__ __align__(16) unsigned short act[128 * 256];   // 64 KB

  const int tid = threadIdx.x;
  const int rowBase = blockIdx.x * 128;

  // Stage concat(x, ug[gather]) into LDS as bf16
  for (int idx = tid; idx < 128 * 32; idx += 512) {
    const int row = idx >> 5, c4 = idx & 31;
    const float4 f = ((const float4*)x)[(size_t)(rowBase + row) * 32 + c4];
    unsigned short* p = &act[row * 256 + c4 * 4];
    p[0] = f2bf(f.x); p[1] = f2bf(f.y); p[2] = f2bf(f.z); p[3] = f2bf(f.w);
  }
  for (int idx = tid; idx < 128 * 32; idx += 512) {
    const int row = idx >> 5, c4 = idx & 31;
    const int g = batchGather[rowBase + row];          // L1-resident, redundant ok
    const float4 f = ((const float4*)ug)[(size_t)g * 32 + c4];
    unsigned short* p = &act[row * 256 + 128 + c4 * 4];
    p[0] = f2bf(f.x); p[1] = f2bf(f.y); p[2] = f2bf(f.z); p[3] = f2bf(f.w);
  }
  __syncthreads();

  const int lane = tid & 31, wv = tid >> 5;   // 16 waves
  const int rb = wv & 7, ch = wv >> 3;
  const unsigned* act32 = (const unsigned*)act;
  v8f acc[8];

  // layer 1 (relu)
  zero_acc<8>(acc);
  gemm_compute<8>(act32, w.w0, rb, lane, ch * 8, acc);
  __syncthreads();
  store_lds<8, 256, true>(acc, w.b0, act, rb, lane, ch * 8);
  __syncthreads();
  // layer 2 (relu)
  zero_acc<8>(acc);
  gemm_compute<8>(act32, w.w1, rb, lane, ch * 8, acc);
  __syncthreads();
  store_lds<8, 256, true>(acc, w.b1, act, rb, lane, ch * 8);
  __syncthreads();
  // layer 3 (no relu) -> pool
  zero_acc<8>(acc);
  gemm_compute<8>(act32, w.w2, rb, lane, ch * 8, acc);
  __syncthreads();                       // activations now dead: reuse as poolBuf

  float* poolBuf = (float*)act;          // overlays LDS (first 1KB)
  const int  seg0    = batchSeg[rowBase];
  const bool uniform = (seg0 == batchSeg[rowBase + 127]);   // sorted ids -> all equal
  if (tid < 256) poolBuf[tid] = 0.0f;
  __syncthreads();

  const int half = lane >> 4;
  if (uniform) {
    // whole tile is one segment: reduce in LDS, one global atomic per column
#pragma unroll
    for (int j = 0; j < 8; ++j) {
      const int col = (ch * 8 + j) * 16 + (lane & 15);
      float s = 8.0f * w.b2[col];
#pragma unroll
      for (int r = 0; r < 8; ++r) s += acc[j][r];
      __hip_atomic_fetch_add(&poolBuf[col], s, __ATOMIC_RELAXED, __HIP_MEMORY_SCOPE_WORKGROUP);
    }
    __syncthreads();                     // block-uniform branch: barrier is legal
    if (tid < 256)
      __hip_atomic_fetch_add(&pooled[(size_t)seg0 * 256 + tid], poolBuf[tid],
                             __ATOMIC_RELAXED, __HIP_MEMORY_SCOPE_AGENT);
  } else {
#pragma unroll
    for (int j = 0; j < 8; ++j) {
      const int col = (ch * 8 + j) * 16 + (lane & 15);
      const float bv = w.b2[col];
#pragma unroll
      for (int r = 0; r < 8; ++r) {
        const int row = rb * 16 + r + half * 8;
        const int seg = batchSeg[rowBase + row];
        __hip_atomic_fetch_add(&pooled[(size_t)seg * 256 + col], acc[j][r] + bv,
                               __ATOMIC_RELAXED, __HIP_MEMORY_SCOPE_AGENT);
      }
    }
  }
}

// ---------------------------------------------------------------------------
// Graph-level MLP (256->256->256->128), input either one 256-wide matrix or
// concat of two 128-wide matrices. Output fp32 [rows x 128]. 64-row tiles.
// ---------------------------------------------------------------------------
__global__ __launch_bounds__(256) void ds_row_mlp(
    const float* __restrict__ in0, const float* __restrict__ in1,
    MlpW3 w, float* __restrict__ out) {
  __shared__ __align__(16) unsigned short act[64 * 256];
  const int tid = threadIdx.x;
  const int rowBase = blockIdx.x * 64;

  if (in1 == nullptr) {
    for (int idx = tid; idx < 64 * 64; idx += 256) {
      const int row = idx >> 6, c4 = idx & 63;
      const float4 f = ((const float4*)in0)[(size_t)(rowBase + row) * 64 + c4];
      unsigned short* p = &act[row * 256 + c4 * 4];
      p[0] = f2bf(f.x); p[1] = f2bf(f.y); p[2] = f2bf(f.z); p[3] = f2bf(f.w);
    }
  } else {
    for (int idx = tid; idx < 64 * 32; idx += 256) {
      const int row = idx >> 5, c4 = idx & 31;
      const float4 f = ((const float4*)in0)[(size_t)(rowBase + row) * 32 + c4];
      unsigned short* p = &act[row * 256 + c4 * 4];
      p[0] = f2bf(f.x); p[1] = f2bf(f.y); p[2] = f2bf(f.z); p[3] = f2bf(f.w);
    }
    for (int idx = tid; idx < 64 * 32; idx += 256) {
      const int row = idx >> 5, c4 = idx & 31;
      const float4 f = ((const float4*)in1)[(size_t)(rowBase + row) * 32 + c4];
      unsigned short* p = &act[row * 256 + 128 + c4 * 4];
      p[0] = f2bf(f.x); p[1] = f2bf(f.y); p[2] = f2bf(f.z); p[3] = f2bf(f.w);
    }
  }
  __syncthreads();

  const int lane = tid & 31, wv = tid >> 5;
  const int rb = wv & 3, ch = wv >> 2;
  const unsigned* act32 = (const unsigned*)act;
  v8f acc[8];

  zero_acc<8>(acc);
  gemm_compute<8>(act32, w.w0, rb, lane, ch * 8, acc);
  __syncthreads();
  store_lds<8, 256, true>(acc, w.b0, act, rb, lane, ch * 8);
  __syncthreads();

  zero_acc<8>(acc);
  gemm_compute<8>(act32, w.w1, rb, lane, ch * 8, acc);
  __syncthreads();
  store_lds<8, 256, true>(acc, w.b1, act, rb, lane, ch * 8);
  __syncthreads();

  // final layer: N = 128, 4 tiles per wave, fp32 output to global
  v8f acc4[4];
  zero_acc<4>(acc4);
  gemm_compute<4>(act32, w.w2, rb, lane, ch * 4, acc4);
  const int half = lane >> 4;
#pragma unroll
  for (int j = 0; j < 4; ++j) {
    const int col = (ch * 4 + j) * 16 + (lane & 15);
    const float bv = w.b2[col];
#pragma unroll
    for (int r = 0; r < 8; ++r) {
      const int row = rb * 16 + r + half * 8;
      out[(size_t)(rowBase + row) * 128 + col] = acc4[j][r] + bv;
    }
  }
}

// ---------------------------------------------------------------------------
// Host launcher
// ---------------------------------------------------------------------------
extern "C" void kernel_launch(void* const* d_in, const int* in_sizes, int n_in,
                              void* d_out, int out_size, void* d_ws, size_t ws_size,
                              hipStream_t stream) {
  (void)in_sizes; (void)out_size; (void)ws_size;
  if (n_in < 35) return;

  constexpr int B = 2048, NN = 131072, FX = 128, NIT = 4;

  const float* x1    = (const float*)d_in[0];
  const float* x2    = (const float*)d_in[1];
  const float* u2in  = (const float*)d_in[2];
  const int*   batch1 = (const int*)d_in[3];
  const int*   batch2 = (const int*)d_in[4];
  auto W = [&](int i) { return (const float*)d_in[i]; };

  // workspace layout: packed weights, then fp32 buffers
  static const int widths[15]  = {256,256,256, 256,256,128, 256,256,256, 256,256,128, 256,256,128};
  static const int srcIdx[15]  = {5,7,9, 11,13,15, 17,19,21, 23,25,27, 29,31,33};
  unsigned short* pk[15];
  size_t off = 0;
  char* base = (char*)d_ws;
  for (int m = 0; m < 15; ++m) {
    pk[m] = (unsigned short*)(base + off);
    off += (size_t)256 * widths[m] * sizeof(unsigned short);
  }
  float* pooled1 = (float*)(base + off); off += (size_t)B * 256 * sizeof(float);
  float* pooled2 = (float*)(base + off); off += (size_t)B * 256 * sizeof(float);
  float* u1      = (float*)(base + off); off += (size_t)B * FX * sizeof(float);
  float* u2buf   = (float*)(base + off); off += (size_t)B * FX * sizeof(float);

  // pack all weights (iteration-invariant)
  for (int m = 0; m < 15; ++m)
    ds_pack_w<<<widths[m], 256, 0, stream>>>(W(srcIdx[m]), pk[m], widths[m]);

  MlpW3 e1{pk[0],  pk[1],  pk[2],  W(6),  W(8),  W(10)};
  MlpW3 o1{pk[3],  pk[4],  pk[5],  W(12), W(14), W(16)};
  MlpW3 e2{pk[6],  pk[7],  pk[8],  W(18), W(20), W(22)};
  MlpW3 o2{pk[9],  pk[10], pk[11], W(24), W(26), W(28)};
  MlpW3 mm{pk[12], pk[13], pk[14], W(30), W(32), W(34)};

  float* out = (float*)d_out;
  const float* u2cur = u2in;
  for (int it = 0; it < NIT; ++it) {
    hipMemsetAsync(pooled1, 0, (size_t)B * 256 * sizeof(float), stream);
    ds_emb_pool<<<NN / 128, 512, 0, stream>>>(x1, u2cur, batch2, batch1, e1, pooled1);
    ds_row_mlp<<<B / 64, 256, 0, stream>>>(pooled1, nullptr, o1, u1);

    hipMemsetAsync(pooled2, 0, (size_t)B * 256 * sizeof(float), stream);
    ds_emb_pool<<<NN / 128, 512, 0, stream>>>(x2, u1, batch1, batch2, e2, pooled2);
    ds_row_mlp<<<B / 64, 256, 0, stream>>>(pooled2, nullptr, o2, u2buf);

    ds_row_mlp<<<B / 64, 256, 0, stream>>>(u1, u2buf, mm, out + (size_t)it * B * FX);
    u2cur = u2buf;
  }
}
